// EnhancedBiLSTM_CRF_49022756717091
// MI455X (gfx1250) — matrix operand order repaired and checked by reading, not compiled, not run on hardware
//
#include <hip/hip_runtime.h>
#include <hip/hip_bf16.h>
#include <math.h>

typedef __attribute__((ext_vector_type(16))) _Float16 v16h;
typedef __attribute__((ext_vector_type(8)))  _Float16 v8h;
typedef __attribute__((ext_vector_type(8)))  float    v8f;

#define S_LEN 4096
#define D_EMB 256
#define HID   512
#define H_DIR 256
#define G4N   2048      // 2 dirs * 4 gates * H
#define A_ATT 128
#define M1DIM 256       // HID/2
#define T_TAG 12
#define T_PAD 16
#define START_TAG 10
#define STOP_TAG  11
#define NEG_VAL  -10000.0f

// ---------------------------------------------------------------------------
// generic f32 -> f16 convert
__global__ void k_cvt_f16(const float* __restrict__ x, _Float16* __restrict__ y, int n) {
    int i = blockIdx.x * blockDim.x + threadIdx.x;
    if (i < n) y[i] = (_Float16)x[i];
}

// embedding gather -> f16 activations
__global__ void k_gather(const int* __restrict__ sent, const float* __restrict__ embed,
                         _Float16* __restrict__ x16, int n) {
    int i = blockIdx.x * blockDim.x + threadIdx.x;
    if (i < n) {
        int s = i >> 8;            // / D_EMB
        int d = i & (D_EMB - 1);
        x16[i] = (_Float16)embed[(size_t)sent[s] * D_EMB + d];
    }
}

// pad W2 (12x256) -> (16x256) f16, b2 (12) -> (16) f32
__global__ void k_pad_w2(const float* __restrict__ W2, const float* __restrict__ b2,
                         _Float16* __restrict__ W2p, float* __restrict__ b2p) {
    int i = blockIdx.x * blockDim.x + threadIdx.x;
    if (i < T_PAD * M1DIM) {
        int r = i / M1DIM, k = i % M1DIM;
        W2p[i] = (r < T_TAG) ? (_Float16)W2[r * M1DIM + k] : (_Float16)0.0f;
    }
    if (i < T_PAD) b2p[i] = (i < T_TAG) ? b2[i] : 0.0f;
}

// ---------------------------------------------------------------------------
// Fragment helpers (16-bit WMMA operands; contiguous k order, same for A and B).
// A (16x32): lane L holds row L&15, K in {base..base+15}, base=16*(L>=16)
// B (32x16): lane L holds col L&15, K in {base..base+15}, base=16*(L>=16)
__device__ __forceinline__ v16h load_a_frag(const _Float16* __restrict__ row, int ak) {
    v8h lo = *reinterpret_cast<const v8h*>(row + ak);
    v8h hi = *reinterpret_cast<const v8h*>(row + ak + 8);
    v16h a;
#pragma unroll
    for (int e = 0; e < 8; ++e) { a[e] = lo[e]; a[e + 8] = hi[e]; }
    return a;
}

// ---------------------------------------------------------------------------
// Register-blocked WMMA GEMM: each wave computes a 64x32 block (4x2 tiles of 16x16).
// Y[M x N] = act( X[M x K] @ W[N x K]^T + bias ); f16 in, f32 accum.
// Requires M % 64 == 0, N % 32 == 0, K % 32 == 0. act: 0=none 1=relu 2=tanh.
__global__ void k_wmma_gemm_b(const _Float16* __restrict__ X,
                              const _Float16* __restrict__ W,
                              const float* __restrict__ bias,
                              float* __restrict__ Y,
                              _Float16* __restrict__ Yh,
                              int M, int N, int K, int act) {
    const int lane = threadIdx.x & 31;
    const int wave = threadIdx.x >> 5;
    const int nb = N >> 5;
    const int blk = blockIdx.x * (blockDim.x >> 5) + wave;
    if (blk >= (M >> 6) * nb) return;               // uniform per wave: EXEC stays all-1

    const int bm = (blk / nb) << 6;                 // 64-row block
    const int bn = (blk % nb) << 5;                 // 32-col block

    const int am = lane & 15;
    const int ak = (lane >> 4) << 4;                // 0 or 16
    const int bnn = lane & 15;
    const int bk = (lane >> 4) << 4;                // 0 or 16

    const _Float16* __restrict__ Xr0 = X + (size_t)(bm +  0 + am) * K;
    const _Float16* __restrict__ Xr1 = X + (size_t)(bm + 16 + am) * K;
    const _Float16* __restrict__ Xr2 = X + (size_t)(bm + 32 + am) * K;
    const _Float16* __restrict__ Xr3 = X + (size_t)(bm + 48 + am) * K;
    const _Float16* __restrict__ Wr0 = W + (size_t)(bn +  0 + bnn) * K;
    const _Float16* __restrict__ Wr1 = W + (size_t)(bn + 16 + bnn) * K;

    v8f c[4][2] = {};
    for (int k0 = 0; k0 < K; k0 += 32) {
        v16h a0 = load_a_frag(Xr0 + k0, ak);
        v16h a1 = load_a_frag(Xr1 + k0, ak);
        v16h a2 = load_a_frag(Xr2 + k0, ak);
        v16h a3 = load_a_frag(Xr3 + k0, ak);
        v16h b0 = *reinterpret_cast<const v16h*>(Wr0 + k0 + bk);
        v16h b1 = *reinterpret_cast<const v16h*>(Wr1 + k0 + bk);
        c[0][0] = __builtin_amdgcn_wmma_f32_16x16x32_f16(false, a0, false, b0, (short)0, c[0][0], false, false);
        c[0][1] = __builtin_amdgcn_wmma_f32_16x16x32_f16(false, a0, false, b1, (short)0, c[0][1], false, false);
        c[1][0] = __builtin_amdgcn_wmma_f32_16x16x32_f16(false, a1, false, b0, (short)0, c[1][0], false, false);
        c[1][1] = __builtin_amdgcn_wmma_f32_16x16x32_f16(false, a1, false, b1, (short)0, c[1][1], false, false);
        c[2][0] = __builtin_amdgcn_wmma_f32_16x16x32_f16(false, a2, false, b0, (short)0, c[2][0], false, false);
        c[2][1] = __builtin_amdgcn_wmma_f32_16x16x32_f16(false, a2, false, b1, (short)0, c[2][1], false, false);
        c[3][0] = __builtin_amdgcn_wmma_f32_16x16x32_f16(false, a3, false, b0, (short)0, c[3][0], false, false);
        c[3][1] = __builtin_amdgcn_wmma_f32_16x16x32_f16(false, a3, false, b1, (short)0, c[3][1], false, false);
    }

    // D layout: VGPR r -> row r + 8*(lane>=16), col = lane&15
    const int rb = (lane >> 4) << 3;
#pragma unroll
    for (int j = 0; j < 2; ++j) {
        const int col = bn + 16 * j + (lane & 15);
        const float bv = bias ? bias[col] : 0.0f;
#pragma unroll
        for (int i = 0; i < 4; ++i) {
#pragma unroll
            for (int r = 0; r < 8; ++r) {
                float v = c[i][j][r] + bv;
                if (act == 1) v = fmaxf(v, 0.0f);
                else if (act == 2) v = tanhf(v);
                const size_t idx = (size_t)(bm + 16 * i + rb + r) * N + col;
                Y[idx] = v;
                if (Yh) Yh[idx] = (_Float16)v;
            }
        }
    }
}

// ---------------------------------------------------------------------------
// Single-tile WMMA GEMM (for small N, e.g. padded tagset N=16).
__global__ void k_wmma_gemm(const _Float16* __restrict__ X,
                            const _Float16* __restrict__ W,
                            const float* __restrict__ bias,
                            float* __restrict__ Y,
                            int M, int N, int K, int act) {
    const int lane = threadIdx.x & 31;
    const int wave = threadIdx.x >> 5;
    const int tilesN = N >> 4;
    const int tile = blockIdx.x * (blockDim.x >> 5) + wave;
    if (tile >= (M >> 4) * tilesN) return;

    const int tm = (tile / tilesN) << 4;
    const int tn = (tile % tilesN) << 4;
    const int am = lane & 15;
    const int ak = (lane >> 4) << 4;
    const int bn = lane & 15;
    const int bk = (lane >> 4) << 4;

    const _Float16* __restrict__ Xrow = X + (size_t)(tm + am) * K;
    const _Float16* __restrict__ Wrow = W + (size_t)(tn + bn) * K;

    v8f c = {};
    for (int k0 = 0; k0 < K; k0 += 32) {
        v16h a = load_a_frag(Xrow + k0, ak);
        v16h b = *reinterpret_cast<const v16h*>(Wrow + k0 + bk);
        c = __builtin_amdgcn_wmma_f32_16x16x32_f16(false, a, false, b, (short)0, c, false, false);
    }

    const int col = tn + (lane & 15);
    const float bv = bias ? bias[col] : 0.0f;
    const int rbase = tm + ((lane >> 4) << 3);
#pragma unroll
    for (int r = 0; r < 8; ++r) {
        float v = c[r] + bv;
        if (act == 1) v = fmaxf(v, 0.0f);
        else if (act == 2) v = tanhf(v);
        Y[(size_t)(rbase + r) * N + col] = v;
    }
}

// ---------------------------------------------------------------------------
// LSTM recurrence: one workgroup per direction, 1024 threads (one gate row each).
// G: S x 2048 preactivations (x@Wih^T + b), Whh: 2 x 1024 x 256 f32.
__global__ void k_lstm_scan(const float* __restrict__ G,
                            const float* __restrict__ Whh,
                            float* __restrict__ hcat, int S) {
    const int dir = blockIdx.x;
    const int tid = threadIdx.x;                    // 0..1023
    const float* __restrict__ Wh = Whh + (size_t)dir * 1024 * H_DIR + (size_t)tid * H_DIR;
    const float4* __restrict__ w4 = reinterpret_cast<const float4*>(Wh);

    __shared__ __align__(16) float hbuf[H_DIR];
    __shared__ float gates[1024];
    if (tid < H_DIR) hbuf[tid] = 0.0f;
    float cst = 0.0f;                               // cell state, owned by tid < 256
    __syncthreads();

    const float4* __restrict__ h4 = reinterpret_cast<const float4*>(hbuf);
    for (int step = 0; step < S; ++step) {
        const int t = dir ? (S - 1 - step) : step;
        float acc = G[(size_t)t * G4N + dir * 1024 + tid];
#pragma unroll 8
        for (int k = 0; k < H_DIR / 4; ++k) {
            float4 w = w4[k], h = h4[k];
            acc += w.x * h.x + w.y * h.y + w.z * h.z + w.w * h.w;
        }
        gates[tid] = acc;
        if (step + 1 < S) {                         // gfx1250 global prefetch of next gate row
            const int tn = dir ? (t - 1) : (t + 1);
            __builtin_prefetch(&G[(size_t)tn * G4N + dir * 1024 + tid], 0, 1);
        }
        __syncthreads();
        if (tid < H_DIR) {
            const float ig = 1.0f / (1.0f + expf(-gates[tid]));
            const float fg = 1.0f / (1.0f + expf(-gates[H_DIR + tid]));
            const float gg = tanhf(gates[2 * H_DIR + tid]);
            const float og = 1.0f / (1.0f + expf(-gates[3 * H_DIR + tid]));
            cst = fg * cst + ig * gg;
            const float h = og * tanhf(cst);
            hbuf[tid] = h;
            hcat[(size_t)t * HID + dir * H_DIR + tid] = h;
        }
        __syncthreads();
    }
}

// ---------------------------------------------------------------------------
// attention score: scores[s] = dot(tanh_att[s][:], v_ctx)
__global__ void k_att_score(const float* __restrict__ att, const float* __restrict__ vctx,
                            float* __restrict__ scores, int S) {
    int s = blockIdx.x * blockDim.x + threadIdx.x;
    if (s < S) {
        const float* row = att + (size_t)s * A_ATT;
        float acc = 0.0f;
#pragma unroll 8
        for (int a = 0; a < A_ATT; ++a) acc += row[a] * vctx[a];
        scores[s] = acc;
    }
}

// single-workgroup softmax stats (deterministic tree reduce): sm[0]=max, sm[1]=1/sum
__global__ void k_softmax_stats(const float* __restrict__ scores, float* __restrict__ sm, int S) {
    __shared__ float red[1024];
    const int tid = threadIdx.x;
    float m = -3.4e38f;
    for (int i = tid; i < S; i += 1024) m = fmaxf(m, scores[i]);
    red[tid] = m; __syncthreads();
    for (int s2 = 512; s2 > 0; s2 >>= 1) { if (tid < s2) red[tid] = fmaxf(red[tid], red[tid + s2]); __syncthreads(); }
    const float gmax = red[0]; __syncthreads();
    float sum = 0.0f;
    for (int i = tid; i < S; i += 1024) sum += expf(scores[i] - gmax);
    red[tid] = sum; __syncthreads();
    for (int s2 = 512; s2 > 0; s2 >>= 1) { if (tid < s2) red[tid] += red[tid + s2]; __syncthreads(); }
    if (tid == 0) { sm[0] = gmax; sm[1] = 1.0f / red[0]; }
}

// hw16[s][d] = (f16)( h[s][d] * softmax(scores)[s] )
__global__ void k_apply_w(const float* __restrict__ h, const float* __restrict__ scores,
                          const float* __restrict__ sm, _Float16* __restrict__ hw16, int n) {
    int i = blockIdx.x * blockDim.x + threadIdx.x;
    if (i < n) {
        int s = i >> 9;                             // / HID
        float w = expf(scores[s] - sm[0]) * sm[1];
        hw16[i] = (_Float16)(h[i] * w);
    }
}

// gold score: deterministic single-workgroup reduction
__global__ void k_gold(const float* __restrict__ feats, const float* __restrict__ trans,
                       const int* __restrict__ tags, float* __restrict__ gold, int S) {
    __shared__ float red[1024];
    const int tid = threadIdx.x;
    float acc = 0.0f;
    for (int t = tid; t < S; t += 1024) {
        const int cur  = tags[t];
        const int prev = (t == 0) ? START_TAG : tags[t - 1];
        acc += feats[(size_t)t * T_PAD + cur] + trans[cur * T_TAG + prev];
    }
    red[tid] = acc; __syncthreads();
    for (int s2 = 512; s2 > 0; s2 >>= 1) { if (tid < s2) red[tid] += red[tid + s2]; __syncthreads(); }
    if (tid == 0) gold[0] = red[0] + trans[STOP_TAG * T_TAG + tags[S - 1]];
}

// CRF forward (log partition) over S steps; single wave; writes final NLL
__global__ void k_crf(const float* __restrict__ feats, const float* __restrict__ trans,
                      const float* __restrict__ gold, float* __restrict__ out, int S) {
    __shared__ float fv[T_TAG];
    const int lane = threadIdx.x;
    float trow[T_TAG];
    if (lane < T_TAG) {
#pragma unroll
        for (int j = 0; j < T_TAG; ++j) trow[j] = trans[lane * T_TAG + j];
        fv[lane] = (lane == START_TAG) ? 0.0f : NEG_VAL;
    }
    __syncthreads();
    for (int t = 0; t < S; ++t) {
        float nf = 0.0f;
        if (lane < T_TAG) {
            float m = -3.4e38f;
#pragma unroll
            for (int j = 0; j < T_TAG; ++j) m = fmaxf(m, fv[j] + trow[j]);
            float sum = 0.0f;
#pragma unroll
            for (int j = 0; j < T_TAG; ++j) sum += expf(fv[j] + trow[j] - m);
            nf = m + logf(sum) + feats[(size_t)t * T_PAD + lane];
        }
        __syncthreads();
        if (lane < T_TAG) fv[lane] = nf;
        __syncthreads();
    }
    if (lane == 0) {
        float m = -3.4e38f;
        for (int j = 0; j < T_TAG; ++j) m = fmaxf(m, fv[j] + trans[STOP_TAG * T_TAG + j]);
        float sum = 0.0f;
        for (int j = 0; j < T_TAG; ++j) sum += expf(fv[j] + trans[STOP_TAG * T_TAG + j] - m);
        out[0] = (m + logf(sum)) - gold[0];
    }
}

// ---------------------------------------------------------------------------
extern "C" void kernel_launch(void* const* d_in, const int* in_sizes, int n_in,
                              void* d_out, int out_size, void* d_ws, size_t ws_size,
                              hipStream_t stream) {
    (void)in_sizes; (void)n_in; (void)out_size; (void)ws_size;
    const int*   sent  = (const int*)  d_in[0];
    const int*   tags  = (const int*)  d_in[1];
    const float* embed = (const float*)d_in[2];
    const float* w0ih  = (const float*)d_in[3];
    const float* w0hh  = (const float*)d_in[4];
    const float* b0    = (const float*)d_in[5];
    const float* w1ih  = (const float*)d_in[6];
    const float* w1hh  = (const float*)d_in[7];
    const float* b1l   = (const float*)d_in[8];
    const float* Wa    = (const float*)d_in[9];
    const float* ba    = (const float*)d_in[10];
    const float* vctx  = (const float*)d_in[11];
    const float* W1    = (const float*)d_in[12];
    const float* b1m   = (const float*)d_in[13];
    const float* W2    = (const float*)d_in[14];
    const float* b2m   = (const float*)d_in[15];
    const float* trans = (const float*)d_in[16];
    float* out = (float*)d_out;

    // workspace carve-up (256B aligned)
    char* ws = (char*)d_ws;
    size_t off = 0;
    auto carve = [&](size_t bytes) { char* p = ws + off; off += (bytes + 255) & ~(size_t)255; return p; };
    _Float16* x16   = (_Float16*)carve((size_t)S_LEN * D_EMB * 2);
    _Float16* wi0h  = (_Float16*)carve((size_t)2 * 1024 * D_EMB * 2);
    _Float16* wi1h  = (_Float16*)carve((size_t)2 * 1024 * HID * 2);
    _Float16* wah   = (_Float16*)carve((size_t)A_ATT * HID * 2);
    _Float16* w1h   = (_Float16*)carve((size_t)M1DIM * HID * 2);
    _Float16* w2p   = (_Float16*)carve((size_t)T_PAD * M1DIM * 2);
    float*    b2p   = (float*)   carve((size_t)T_PAD * 4);
    float*    G     = (float*)   carve((size_t)S_LEN * G4N * 4);      // both layers reuse
    float*    hcat  = (float*)   carve((size_t)S_LEN * HID * 4);      // both layers reuse
    _Float16* h16   = (_Float16*)carve((size_t)S_LEN * HID * 2);      // h0, h1, then weighted-h
    float*    att   = (float*)   carve((size_t)S_LEN * A_ATT * 4);
    float*    scr   = (float*)   carve((size_t)S_LEN * 4);
    float*    sm    = (float*)   carve(256);
    float*    mh    = (float*)   carve((size_t)S_LEN * M1DIM * 4);
    _Float16* mh16  = (_Float16*)carve((size_t)S_LEN * M1DIM * 2);
    float*    feats = (float*)   carve((size_t)S_LEN * T_PAD * 4);
    float*    gold  = (float*)   carve(256);

    const int TB = 256;
    auto blocks = [](int n, int tb) { return (n + tb - 1) / tb; };

    // --- weight prep (f32 -> f16) ---
    k_cvt_f16<<<blocks(2 * 1024 * D_EMB, TB), TB, 0, stream>>>(w0ih, wi0h, 2 * 1024 * D_EMB);
    k_cvt_f16<<<blocks(2 * 1024 * HID,  TB), TB, 0, stream>>>(w1ih, wi1h, 2 * 1024 * HID);
    k_cvt_f16<<<blocks(A_ATT * HID,     TB), TB, 0, stream>>>(Wa,   wah,  A_ATT * HID);
    k_cvt_f16<<<blocks(M1DIM * HID,     TB), TB, 0, stream>>>(W1,   w1h,  M1DIM * HID);
    k_pad_w2<<<blocks(T_PAD * M1DIM,    TB), TB, 0, stream>>>(W2, b2m, w2p, b2p);

    // --- embedding gather ---
    k_gather<<<blocks(S_LEN * D_EMB, TB), TB, 0, stream>>>(sent, embed, x16, S_LEN * D_EMB);

    // --- layer 0: input projection GEMM (N=2048 fuses both dirs + 4 gates), then scan ---
    {
        const int wblk = (S_LEN / 64) * (G4N / 32);
        k_wmma_gemm_b<<<blocks(wblk, 8), 256, 0, stream>>>(x16, wi0h, b0, G, nullptr,
                                                           S_LEN, G4N, D_EMB, 0);
    }
    k_lstm_scan<<<2, 1024, 0, stream>>>(G, w0hh, hcat, S_LEN);
    k_cvt_f16<<<blocks(S_LEN * HID, TB), TB, 0, stream>>>(hcat, h16, S_LEN * HID);

    // --- layer 1 ---
    {
        const int wblk = (S_LEN / 64) * (G4N / 32);
        k_wmma_gemm_b<<<blocks(wblk, 8), 256, 0, stream>>>(h16, wi1h, b1l, G, nullptr,
                                                           S_LEN, G4N, HID, 0);
    }
    k_lstm_scan<<<2, 1024, 0, stream>>>(G, w1hh, hcat, S_LEN);
    k_cvt_f16<<<blocks(S_LEN * HID, TB), TB, 0, stream>>>(hcat, h16, S_LEN * HID);

    // --- attention: tanh(h@Wa^T + ba) via WMMA, dot with v_ctx, softmax over seq ---
    {
        const int wblk = (S_LEN / 64) * (A_ATT / 32);
        k_wmma_gemm_b<<<blocks(wblk, 8), 256, 0, stream>>>(h16, wah, ba, att, nullptr,
                                                           S_LEN, A_ATT, HID, 2);
    }
    k_att_score<<<blocks(S_LEN, TB), TB, 0, stream>>>(att, vctx, scr, S_LEN);
    k_softmax_stats<<<1, 1024, 0, stream>>>(scr, sm, S_LEN);
    k_apply_w<<<blocks(S_LEN * HID, TB), TB, 0, stream>>>(hcat, scr, sm, h16, S_LEN * HID);

    // --- MLP head: relu(hw@W1^T + b1) @ W2p^T + b2p ---
    {
        const int wblk = (S_LEN / 64) * (M1DIM / 32);
        k_wmma_gemm_b<<<blocks(wblk, 8), 256, 0, stream>>>(h16, w1h, b1m, mh, mh16,
                                                           S_LEN, M1DIM, HID, 1);
    }
    {
        const int tiles = (S_LEN / 16) * (T_PAD / 16);
        k_wmma_gemm<<<blocks(tiles, 8), 256, 0, stream>>>(mh16, w2p, b2p, feats,
                                                          S_LEN, T_PAD, M1DIM, 0);
    }

    // --- gold score + CRF forward -> NLL ---
    k_gold<<<1, 1024, 0, stream>>>(feats, trans, tags, gold, S_LEN);
    k_crf<<<1, 32, 0, stream>>>(feats, trans, gold, out, S_LEN);
}
